// GQAAttention_46033459479122
// MI455X (gfx1250) — matrix operand order repaired and checked by reading, not compiled, not run on hardware
//
#include <hip/hip_runtime.h>
#include <hip/hip_bf16.h>

#define QL    2048
#define HISTC 2048
#define KL    4096      // HISTC + QL
#define NH    12
#define NKVH  2
#define HD    128
#define HIDC  1536
#define GRP   6         // NH / NKVH

typedef __attribute__((ext_vector_type(16))) _Float16 v16h;
typedef __attribute__((ext_vector_type(8)))  _Float16 v8h;
typedef __attribute__((ext_vector_type(8)))  float    v8f;

static __device__ __forceinline__ v16h join8(v8h a, v8h b) {
  return __builtin_shufflevector(a, b, 0,1,2,3,4,5,6,7,8,9,10,11,12,13,14,15);
}
static __device__ __forceinline__ v8f wmma_f16(v16h a, v16h b, v8f c) {
  // D = A(16x32 f16) x B(32x16 f16) + C(16x16 f32)
  return __builtin_amdgcn_wmma_f32_16x16x32_f16(false, a, false, b, (short)0, c,
                                                false, false);
}

// ---------------------------------------------------------------- elementwise
__global__ __launch_bounds__(256) void gqa_cvt_f16(const float* __restrict__ in,
                                                   _Float16* __restrict__ out, int n) {
  int i = blockIdx.x * 256 + threadIdx.x;
  if (i < n) out[i] = (_Float16)in[i];
}

// in: f32 [K,N] row-major -> out: f16 [N,K] row-major (B^T for WMMA B fragments)
__global__ __launch_bounds__(256) void gqa_transpose_f16(const float* __restrict__ in,
                                                         _Float16* __restrict__ out,
                                                         int K, int N) {
  int i = blockIdx.x * 256 + threadIdx.x;
  if (i >= K * N) return;
  int k = i / N, n = i - k * N;
  out[(size_t)n * K + k] = (_Float16)in[i];
}

// k_hist/v_hist f32 [HISTC, NKVH, HD] -> Kc f16 [NKVH, KL, HD], Vt f16 [NKVH, HD, KL]
__global__ __launch_bounds__(256) void gqa_hist(const float* __restrict__ kh,
                                                const float* __restrict__ vh,
                                                _Float16* __restrict__ Kc,
                                                _Float16* __restrict__ Vt) {
  int i = blockIdx.x * 256 + threadIdx.x;
  if (i >= HISTC * NKVH * HD) return;
  int t = i / (NKVH * HD);
  int r = i - t * (NKVH * HD);
  int kvh = r / HD;
  int d = r - kvh * HD;
  Kc[((size_t)kvh * KL + t) * HD + d] = (_Float16)kh[i];
  Vt[((size_t)kvh * HD + d) * KL + t] = (_Float16)vh[i];
}

// Qf f32 [QL, HIDC] -> rope -> Qh f16 [NH, QL, HD]
__global__ __launch_bounds__(256) void gqa_rope_q(const float* __restrict__ Qf,
                                                  const float* __restrict__ cosx,
                                                  const float* __restrict__ sinx,
                                                  _Float16* __restrict__ Qh) {
  int i = blockIdx.x * 256 + threadIdx.x;
  if (i >= QL * NH * 64) return;
  int q = i / (NH * 64);
  int r = i - q * (NH * 64);
  int h = r / 64;
  int d = r - h * 64;
  float x1 = Qf[(size_t)q * HIDC + h * HD + d];
  float x2 = Qf[(size_t)q * HIDC + h * HD + d + 64];
  float c = cosx[q * 64 + d], s = sinx[q * 64 + d];
  _Float16* o = Qh + ((size_t)h * QL + q) * HD;
  o[d]      = (_Float16)(x1 * c - x2 * s);
  o[d + 64] = (_Float16)(x1 * s + x2 * c);
}

// Kf f32 [QL, NKVH*HD] -> rope -> Kc f16 [NKVH, KL, HD] at offset HISTC
__global__ __launch_bounds__(256) void gqa_rope_k(const float* __restrict__ Kf,
                                                  const float* __restrict__ cosx,
                                                  const float* __restrict__ sinx,
                                                  _Float16* __restrict__ Kc) {
  int i = blockIdx.x * 256 + threadIdx.x;
  if (i >= QL * NKVH * 64) return;
  int q = i / (NKVH * 64);
  int r = i - q * (NKVH * 64);
  int kvh = r / 64;
  int d = r - kvh * 64;
  float x1 = Kf[(size_t)q * (NKVH * HD) + kvh * HD + d];
  float x2 = Kf[(size_t)q * (NKVH * HD) + kvh * HD + d + 64];
  float c = cosx[q * 64 + d], s = sinx[q * 64 + d];
  _Float16* o = Kc + ((size_t)kvh * KL + HISTC + q) * HD;
  o[d]      = (_Float16)(x1 * c - x2 * s);
  o[d + 64] = (_Float16)(x1 * s + x2 * c);
}

// Vf f32 [QL, NKVH*HD] -> Vt f16 [NKVH, HD, KL] at offset HISTC (transposed scatter)
__global__ __launch_bounds__(256) void gqa_scatter_v(const float* __restrict__ Vf,
                                                     _Float16* __restrict__ Vt) {
  int i = blockIdx.x * 256 + threadIdx.x;
  if (i >= QL * NKVH * HD) return;
  int q = i / (NKVH * HD);
  int r = i - q * (NKVH * HD);
  int kvh = r / HD;
  int d = r - kvh * HD;
  Vt[((size_t)kvh * HD + d) * KL + HISTC + q] = (_Float16)Vf[i];
}

// ---------------------------------------------------------------- WMMA GEMM
// C[M,N](f32) = A[M,K](f16 row-major) @ B (as BT[N,K] f16 row-major) + bias.
// One wave computes a 32x64 tile (2 M-tiles x 4 N-tiles) -> 12 b128 loads / 8 WMMA.
__global__ __launch_bounds__(256) void gqa_gemm(const _Float16* __restrict__ A,
                                                const _Float16* __restrict__ BT,
                                                const float* __restrict__ bias,
                                                float* __restrict__ C,
                                                int M, int N, int K) {
  int wave = (int)((blockIdx.x * 256u + threadIdx.x) >> 5);
  int lane = threadIdx.x & 31;
  int nblk = N >> 6;                       // N / 64
  int mb = wave / nblk;
  int nb = wave - mb * nblk;
  if (mb >= (M >> 5)) return;              // wave-uniform (EXEC all-1 for WMMA)
  int li = lane & 15, hi = lane >> 4;

  const _Float16* ar0 = A + (size_t)(mb * 32 + li) * K;
  const _Float16* ar1 = ar0 + (size_t)16 * K;
  const _Float16* br[4];
#pragma unroll
  for (int j = 0; j < 4; ++j) br[j] = BT + (size_t)(nb * 64 + j * 16 + li) * K;

  v8f zero = {};
  v8f acc[2][4];
#pragma unroll
  for (int i = 0; i < 2; ++i)
#pragma unroll
    for (int j = 0; j < 4; ++j) acc[i][j] = zero;

  for (int k = 0; k < K; k += 32) {
    if (k + 64 < K) {
      __builtin_prefetch(ar0 + k + 64, 0, 0);   // global_prefetch_b8
      __builtin_prefetch(ar1 + k + 64, 0, 0);
    }
    v16h a0 = join8(*(const v8h*)(ar0 + k + hi * 16),
                    *(const v8h*)(ar0 + k + hi * 16 + 8));
    v16h a1 = join8(*(const v8h*)(ar1 + k + hi * 16),
                    *(const v8h*)(ar1 + k + hi * 16 + 8));
#pragma unroll
    for (int j = 0; j < 4; ++j) {
      v16h b = *(const v16h*)(br[j] + k + hi * 16);
      acc[0][j] = wmma_f16(a0, b, acc[0][j]);
      acc[1][j] = wmma_f16(a1, b, acc[1][j]);
    }
  }

#pragma unroll
  for (int i = 0; i < 2; ++i)
#pragma unroll
    for (int j = 0; j < 4; ++j) {
      int col = nb * 64 + j * 16 + li;
      float bv = bias ? bias[col] : 0.0f;
      float* crow = C + (size_t)(mb * 32 + i * 16 + hi * 8) * N + col;
#pragma unroll
      for (int r = 0; r < 8; ++r) crow[(size_t)r * N] = acc[i][j][r] + bv;
    }
}

// ---------------------------------------------------------------- attention
// Block = 4 waves = 64 queries of one head. K/V 32-key blocks are cooperatively
// staged into LDS double buffers (global->reg loads for block i+1 overlap
// compute on block i); each wave runs flash online-softmax on its 16-q tile.
__global__ __launch_bounds__(128) void gqa_attn(const _Float16* __restrict__ Qh,  // [NH,QL,HD]
                                                const _Float16* __restrict__ Kc,  // [NKVH,KL,HD]
                                                const _Float16* __restrict__ Vt,  // [NKVH,HD,KL]
                                                _Float16* __restrict__ O) {      // [QL,HIDC]
  __shared__ _Float16 Kbuf[2][32 * HD];    // 2 x 8KB : keys x d
  __shared__ _Float16 Vbuf[2][HD * 32];    // 2 x 8KB : d x keys
  __shared__ _Float16 plds[4][16 * 32];    // per-wave P staging

  const int t    = threadIdx.x;            // 0..127
  const int lane = t & 31;
  const int wv   = t >> 5;
  const int h    = blockIdx.x >> 5;        // 32 q-blocks per head
  const int qblk = blockIdx.x & 31;
  const int kvh  = h / GRP;
  const int qb0  = qblk * 64;
  const int q0   = qb0 + wv * 16;
  const int li   = lane & 15, hi = lane >> 4;

  const _Float16* kbp = Kc + (size_t)kvh * KL * HD;
  const _Float16* vbp = Vt + (size_t)kvh * HD * KL;

  // Q tile 16x128 as four A fragments (reused across all key blocks)
  const _Float16* qbase = Qh + ((size_t)h * QL + q0 + li) * HD;
  v16h qa[4];
#pragma unroll
  for (int f = 0; f < 4; ++f) {
    const _Float16* p = qbase + f * 32 + hi * 16;
    qa[f] = join8(*(const v8h*)p, *(const v8h*)(p + 8));
  }

  v8f zero = {};
  v8f acc[8];
#pragma unroll
  for (int j = 0; j < 8; ++j) acc[j] = zero;
  float m_i[8], l_i[8];
#pragma unroll
  for (int r = 0; r < 8; ++r) { m_i[r] = -3.0e38f; l_i[r] = 0.0f; }

  const float scale = 0.08838834764831845f;   // 1/sqrt(128)
  const int kend = qb0 + 64 + HISTC;          // uniform over block; <= KL

  v8h kreg[4], vreg[4];
  auto stage_load = [&](int kb) {
#pragma unroll
    for (int i = 0; i < 4; ++i)              // K block is one contiguous 8KB span
      kreg[i] = *(const v8h*)(kbp + (size_t)kb * HD + (t + i * 128) * 8);
#pragma unroll
    for (int i = 0; i < 4; ++i) {            // V^T block: 128 rows x 64B
      int c = t + i * 128;
      int d = c >> 2, off = (c & 3) * 8;
      vreg[i] = *(const v8h*)(vbp + (size_t)d * KL + kb + off);
    }
  };

  stage_load(0);
  int p = 0;
  for (int kb = 0; kb < kend; kb += 32, p ^= 1) {
    // ---- cooperative LDS fill from staged registers
#pragma unroll
    for (int i = 0; i < 4; ++i)
      *(v8h*)(&Kbuf[p][(t + i * 128) * 8]) = kreg[i];
#pragma unroll
    for (int i = 0; i < 4; ++i) {
      int c = t + i * 128;
      int d = c >> 2, off = (c & 3) * 8;
      *(v8h*)(&Vbuf[p][d * 32 + off]) = vreg[i];
    }
    __syncthreads();
    if (kb + 32 < kend) stage_load(kb + 32);  // global loads overlap compute

    // ---- S = Q @ K^T for keys [kb, kb+32), B fragments from LDS
    v8f s0 = zero, s1 = zero;
    const _Float16* kr0 = &Kbuf[p][li * HD];
    const _Float16* kr1 = &Kbuf[p][(16 + li) * HD];
#pragma unroll
    for (int f = 0; f < 4; ++f) {
      int kk = f * 32 + hi * 16;
      s0 = wmma_f16(qa[f], *(const v16h*)(kr0 + kk), s0);
      s1 = wmma_f16(qa[f], *(const v16h*)(kr1 + kk), s1);
    }

    // ---- online softmax (row = q0 + r + hi*8, col = kb + li [+16])
    float p0[8], p1[8];
#pragma unroll
    for (int r = 0; r < 8; ++r) {
      const int lim = q0 + r + hi * 8 + HISTC;      // causal: key <= lim
      float v0 = (kb + li      <= lim) ? s0[r] * scale : -3.0e38f;
      float v1 = (kb + 16 + li <= lim) ? s1[r] * scale : -3.0e38f;
      float mx = fmaxf(v0, v1);
      mx = fmaxf(mx, __shfl_xor(mx, 1, 16));
      mx = fmaxf(mx, __shfl_xor(mx, 2, 16));
      mx = fmaxf(mx, __shfl_xor(mx, 4, 16));
      mx = fmaxf(mx, __shfl_xor(mx, 8, 16));
      float mnew = fmaxf(m_i[r], mx);
      float corr = __expf(m_i[r] - mnew);
      m_i[r] = mnew;
      float e0 = __expf(v0 - mnew);
      float e1 = __expf(v1 - mnew);
      float rs = e0 + e1;
      rs += __shfl_xor(rs, 1, 16);
      rs += __shfl_xor(rs, 2, 16);
      rs += __shfl_xor(rs, 4, 16);
      rs += __shfl_xor(rs, 8, 16);
      l_i[r] = l_i[r] * corr + rs;
      p0[r] = e0; p1[r] = e1;
#pragma unroll
      for (int j = 0; j < 8; ++j) acc[j][r] *= corr;
    }

    // ---- P: C-layout -> A-layout via per-wave LDS (DS in-order per wave)
    _Float16* pl = plds[wv];
#pragma unroll
    for (int r = 0; r < 8; ++r) {
      int row = r + hi * 8;
      pl[row * 32 + li]      = (_Float16)p0[r];
      pl[row * 32 + 16 + li] = (_Float16)p1[r];
    }
    asm volatile("s_wait_dscnt 0" ::: "memory");
    const _Float16* pp = pl + li * 32 + hi * 16;
    v16h pa = join8(*(const v8h*)pp, *(const v8h*)(pp + 8));

    // ---- acc += P(16x32) @ V(32x128), B fragments from LDS V^T
#pragma unroll
    for (int j = 0; j < 8; ++j) {
      v16h bv = *(const v16h*)(&Vbuf[p][(j * 16 + li) * 32 + hi * 16]);
      acc[j] = wmma_f16(pa, bv, acc[j]);
    }
  }

  // ---- epilogue: O[q, h*HD + d] = acc / l
#pragma unroll
  for (int r = 0; r < 8; ++r) {
    float inv = 1.0f / l_i[r];
    _Float16* orow = O + (size_t)(q0 + r + hi * 8) * HIDC + h * HD;
#pragma unroll
    for (int j = 0; j < 8; ++j)
      orow[j * 16 + li] = (_Float16)(acc[j][r] * inv);
  }
}

// ---------------------------------------------------------------- launcher
extern "C" void kernel_launch(void* const* d_in, const int* in_sizes, int n_in,
                              void* d_out, int out_size, void* d_ws, size_t ws_size,
                              hipStream_t stream) {
  const float* x    = (const float*)d_in[0];   // [QL, HIDC]
  const float* Wq   = (const float*)d_in[1];   // [HIDC, HIDC]
  const float* bq   = (const float*)d_in[2];
  const float* Wk   = (const float*)d_in[3];   // [HIDC, 256]
  const float* bk   = (const float*)d_in[4];
  const float* Wv   = (const float*)d_in[5];
  const float* bv   = (const float*)d_in[6];
  const float* Wo   = (const float*)d_in[7];   // [HIDC, HIDC]
  const float* kh   = (const float*)d_in[8];   // [HISTC, NKVH, HD]
  const float* vh   = (const float*)d_in[9];
  const float* fcos = (const float*)d_in[10];  // [QL, 64]
  const float* fsin = (const float*)d_in[11];
  float* out = (float*)d_out;                  // [QL, HIDC] f32

  char* w = (char*)d_ws;
  size_t off = 0;
  auto alloc = [&](size_t bytes) -> void* {
    void* p = w + off;
    off = (off + bytes + 255) & ~(size_t)255;
    return p;
  };
  _Float16* Xh  = (_Float16*)alloc((size_t)QL * HIDC * 2);
  _Float16* WqT = (_Float16*)alloc((size_t)HIDC * HIDC * 2);
  _Float16* WkT = (_Float16*)alloc((size_t)HIDC * 256 * 2);
  _Float16* WvT = (_Float16*)alloc((size_t)HIDC * 256 * 2);
  _Float16* WoT = (_Float16*)alloc((size_t)HIDC * HIDC * 2);
  float*    Qf  = (float*)   alloc((size_t)QL * HIDC * 4);
  float*    Kf  = (float*)   alloc((size_t)QL * 256 * 4);
  float*    Vf  = (float*)   alloc((size_t)QL * 256 * 4);
  _Float16* Qhh = (_Float16*)alloc((size_t)NH * QL * HD * 2);
  _Float16* Kc  = (_Float16*)alloc((size_t)NKVH * KL * HD * 2);
  _Float16* Vt  = (_Float16*)alloc((size_t)NKVH * HD * KL * 2);
  _Float16* Oh  = (_Float16*)alloc((size_t)QL * HIDC * 2);
  if (off > ws_size) return;  // workspace too small; nothing safe to do

  auto blocks = [](int n) { return (n + 255) / 256; };

  // 1) f16 conversions / transposes
  gqa_cvt_f16<<<blocks(QL * HIDC), 256, 0, stream>>>(x, Xh, QL * HIDC);
  gqa_transpose_f16<<<blocks(HIDC * HIDC), 256, 0, stream>>>(Wq, WqT, HIDC, HIDC);
  gqa_transpose_f16<<<blocks(HIDC * 256), 256, 0, stream>>>(Wk, WkT, HIDC, 256);
  gqa_transpose_f16<<<blocks(HIDC * 256), 256, 0, stream>>>(Wv, WvT, HIDC, 256);
  gqa_transpose_f16<<<blocks(HIDC * HIDC), 256, 0, stream>>>(Wo, WoT, HIDC, HIDC);
  gqa_hist<<<blocks(HISTC * NKVH * HD), 256, 0, stream>>>(kh, vh, Kc, Vt);

  // 2) projections (WMMA): waves = (M/32)*(N/64), 8 waves per 256-thread block
  {
    int waves = (QL / 32) * (HIDC / 64);       // 1536
    gqa_gemm<<<waves * 32 / 256, 256, 0, stream>>>(Xh, WqT, bq, Qf, QL, HIDC, HIDC);
  }
  {
    int waves = (QL / 32) * (256 / 64);        // 256
    gqa_gemm<<<waves * 32 / 256, 256, 0, stream>>>(Xh, WkT, bk, Kf, QL, 256, HIDC);
    gqa_gemm<<<waves * 32 / 256, 256, 0, stream>>>(Xh, WvT, bv, Vf, QL, 256, HIDC);
  }

  // 3) RoPE + cache scatter
  gqa_rope_q<<<blocks(QL * NH * 64), 256, 0, stream>>>(Qf, fcos, fsin, Qhh);
  gqa_rope_k<<<blocks(QL * NKVH * 64), 256, 0, stream>>>(Kf, fcos, fsin, Kc);
  gqa_scatter_v<<<blocks(QL * NKVH * HD), 256, 0, stream>>>(Vf, Vt);

  // 4) attention (WMMA flash): 12 heads * 32 q-blocks, 4 waves/block
  gqa_attn<<<NH * (QL / 64), 128, 0, stream>>>(Qhh, Kc, Vt, Oh);

  // 5) output projection (WMMA) -> f32 d_out
  {
    int waves = (QL / 32) * (HIDC / 64);
    gqa_gemm<<<waves * 32 / 256, 256, 0, stream>>>(Oh, WoT, nullptr, out, QL, HIDC, HIDC);
  }
  (void)in_sizes; (void)n_in; (void)out_size;
}
